// MultiAttentionFusion_64450279244499
// MI455X (gfx1250) — matrix operand order repaired and checked by reading, not compiled, not run on hardware
//
#include <hip/hip_runtime.h>

typedef __bf16 bf16;
typedef __attribute__((ext_vector_type(16))) __bf16 v16bf;
typedef __attribute__((ext_vector_type(8)))  __bf16 v8bf;
typedef __attribute__((ext_vector_type(8)))  float  v8f;
typedef __attribute__((ext_vector_type(4)))  unsigned int v4u;
typedef __attribute__((ext_vector_type(8)))  int v8i_;
typedef __attribute__((ext_vector_type(4)))  int v4i_;

#define LDA_SH 40    // 64x32 A tile, padded row stride (elements): 64B row + 16B pad
#define LDB_SH 264   // 32x256 B tile, padded row stride (elements): 512B row + 16B pad

#ifndef __has_builtin
#define __has_builtin(x) 0
#endif

#if defined(__gfx1250__) && __has_builtin(__builtin_amdgcn_tensor_load_to_lds)
#define HAVE_TDM 1
#else
#define HAVE_TDM 0
#endif

#if defined(__gfx1250__) && __has_builtin(__builtin_amdgcn_s_wait_tensorcnt)
#define HAVE_TCNT 1
#else
#define HAVE_TCNT 0
#endif

__device__ __forceinline__ unsigned encodeKey(float f) {
    unsigned u = __float_as_uint(f);
    return (u & 0x80000000u) ? ~u : (u | 0x80000000u);
}
__device__ __forceinline__ float decodeKey(unsigned k) {
    return (k & 0x80000000u) ? __uint_as_float(k ^ 0x80000000u) : __uint_as_float(~k);
}

// Issue a 2-D TDM tile load: tileW x tileH elements (2-byte), row stride strideElems,
// LDS destination padded by (padAmount+1) DWORDs every 2^(padInterval+1) DWORDs.
__device__ __forceinline__ void tdm_load_2d(unsigned ldsByteAddr, const void* gaddr,
                                            unsigned tileW, unsigned tileH,
                                            unsigned strideElems,
                                            unsigned padInterval, unsigned padAmount)
{
#if HAVE_TDM
    unsigned long long ga = (unsigned long long)(size_t)gaddr;
    v4u g0;
    g0[0] = 1u;                                            // count=1, user mode
    g0[1] = ldsByteAddr;                                   // lds_addr
    g0[2] = (unsigned)(ga & 0xFFFFFFFFu);                  // global_addr[31:0]
    g0[3] = (unsigned)((ga >> 32) & 0x01FFFFFFu) | (2u << 30); // addr[56:32] | type=2
    v8i_ g1;
    unsigned w0 = (1u << 16)                               // data_size = 2 bytes
                | (1u << 20)                               // pad_enable
                | (padInterval << 22) | (padAmount << 25);
    g1[0] = (int)w0;
    g1[1] = (int)((tileW & 0xFFFFu) << 16);                // tensor_dim0[15:0]
    g1[2] = (int)((tileW >> 16) | ((tileH & 0xFFFFu) << 16)); // dim0 hi | tensor_dim1 lo
    g1[3] = (int)((tileH >> 16) | (tileW << 16));          // dim1 hi | tile_dim0
    g1[4] = (int)(tileH & 0xFFFFu);                        // tile_dim1 | tile_dim2=0
    g1[5] = (int)strideElems;                              // tensor_dim0_stride lo
    g1[6] = 0;
    g1[7] = 0;
    v4i_ zz = {0, 0, 0, 0};
#if defined(__clang_major__) && (__clang_major__ >= 23)
    v8i_ z8 = {0, 0, 0, 0, 0, 0, 0, 0};
    __builtin_amdgcn_tensor_load_to_lds(g0, g1, zz, zz, z8, 0);
#else
    __builtin_amdgcn_tensor_load_to_lds(g0, g1, zz, zz, 0);
#endif
#else
    (void)ldsByteAddr; (void)gaddr; (void)tileW; (void)tileH;
    (void)strideElems; (void)padInterval; (void)padAmount;
#endif
}

__device__ __forceinline__ void tdm_wait0()
{
#if HAVE_TCNT
    __builtin_amdgcn_s_wait_tensorcnt(0);
#elif HAVE_TDM
    asm volatile("s_wait_tensorcnt 0x0" ::: "memory");
#endif
}

// ---------------------------------------------------------------------------
// Generic bf16 WMMA GEMM, C = A(MxK) * B(KxN), fp32 accum, TDM double-buffered.
// Workgroup tile 64(M) x 256(N); 8 waves in 2x4; wave tile 32x64 (2x4 WMMA tiles).
//   BT=true : B source is N-major (Bsrc[n][k]) -> transposed manual stage
//   GATHER  : A rows indirected through gidx (manual stage)
// Epilogues:
//   EPI=0 : out_bf16 = acc + bias[n]                         (QKV)
//   EPI=1 : out_f32 = acc; global atomic-max of acc          (scores)
//   EPI=2 : out_bf16 = exp(-max(sq[i]+sq[j]-2acc,0)*scale)   (Gaussian W)
//   EPI=3 : out_f32  = nw0*agg[n] + nw1*acc/1024             (final output)
// ---------------------------------------------------------------------------
template<int EPI, bool BT, bool GATHER>
__global__ __launch_bounds__(256)
void gemm_wmma(const bf16* __restrict__ A0, const bf16* __restrict__ B0,
               int lda, int ldb, int K,
               size_t zA, size_t zB,
               const int* __restrict__ gidx,
               const float* __restrict__ bias,
               bf16* __restrict__ outBf0, int ldo, size_t zO,
               float* __restrict__ outF,
               unsigned* __restrict__ gmaxKey,
               const float* __restrict__ sqBase, size_t zSq,
               float scaleExp,
               const float* __restrict__ aggV,
               const float* __restrict__ attnw, const float* __restrict__ noisev,
               float* __restrict__ outFin0, size_t zF)
{
    __shared__ alignas(16) bf16 ldsA[2][64 * LDA_SH];
    __shared__ alignas(16) bf16 ldsB[2][32 * LDB_SH];

    constexpr bool TDMA   = (HAVE_TDM != 0) && !GATHER;
    constexpr bool TDMB   = (HAVE_TDM != 0) && !BT;
    constexpr bool ANYTDM = TDMA || TDMB;

    const int tid   = threadIdx.x;
    const int lane  = tid & 31;
    const int laneN = lane & 15;
    const int laneH = lane >> 4;
    const int wv    = tid >> 5;       // wave id 0..7
    const int mw    = wv >> 2;        // 0..1  (wave M position)
    const int nv    = wv & 3;         // 0..3  (wave N position)
    const int mBase = blockIdx.x * 64;
    const int nBase = blockIdx.y * 256;
    const int z     = blockIdx.z;

    const bf16* A  = A0 + (size_t)z * zA;
    const bf16* Bm = B0 + (size_t)z * zB;
    const float* sq = nullptr;
    if constexpr (EPI == 2) sq = sqBase + (size_t)z * zSq;

    v8f acc[2][4] = {};

    auto stage = [&](int kt, int buf) {
        const int k0 = kt * 32;
        // ---- A tile: 64 rows x 32 k ----
        if constexpr (TDMA) {
            if (wv == 0)
                tdm_load_2d((unsigned)(size_t)(void*)&ldsA[buf][0],
                            (const void*)(A + (size_t)mBase * lda + k0),
                            32u, 64u, (unsigned)lda, 3u, 3u);
        } else {
            int row = tid >> 2, c = (tid & 3) * 8;
            int grow = mBase + row;
            if (GATHER) grow = gidx[grow];
            v8bf va = *(const v8bf*)(A + (size_t)grow * lda + k0 + c);
            *(v8bf*)(&ldsA[buf][row * LDA_SH + c]) = va;
        }
        // ---- B tile into ldsB[k][n] : 32 k x 256 n ----
        if constexpr (BT) {
            #pragma unroll
            for (int cc = 0; cc < 4; cc++) {
                int chunk = tid + cc * 256;          // 1024 chunks of 8 k-elems
                int n = chunk >> 2, k8 = (chunk & 3) * 8;
                v8bf vb = *(const v8bf*)(Bm + (size_t)(nBase + n) * ldb + k0 + k8);
                #pragma unroll
                for (int i = 0; i < 8; i++) ldsB[buf][(k8 + i) * LDB_SH + n] = vb[i];
            }
        } else if constexpr (TDMB) {
            if (wv == 0)
                tdm_load_2d((unsigned)(size_t)(void*)&ldsB[buf][0],
                            (const void*)(Bm + (size_t)k0 * ldb + nBase),
                            256u, 32u, (unsigned)ldb, 6u, 3u);
        } else {
            #pragma unroll
            for (int cc = 0; cc < 4; cc++) {
                int chunk = tid + cc * 256;          // 1024 chunks of 8 n-elems
                int kr = chunk >> 5, nc = (chunk & 31) * 8;
                v8bf vb = *(const v8bf*)(Bm + (size_t)(k0 + kr) * ldb + nBase + nc);
                *(v8bf*)(&ldsB[buf][kr * LDB_SH + nc]) = vb;
            }
        }
    };

    auto tdmWait = [&]() {
        if constexpr (ANYTDM) {
            if (wv == 0) tdm_wait0();
        }
    };

    auto compute = [&](int buf) {
        v16bf af[2], bfr[4];
        #pragma unroll
        for (int ti = 0; ti < 2; ti++) {
            const bf16* p = &ldsA[buf][(mw * 32 + ti * 16 + laneN) * LDA_SH + laneH * 8];
            v8bf lo = *(const v8bf*)p;          // k = laneH*8 + 0..7
            v8bf hi = *(const v8bf*)(p + 16);   // k = 16 + laneH*8 + 0..7
            af[ti] = __builtin_shufflevector(lo, hi, 0,1,2,3,4,5,6,7,8,9,10,11,12,13,14,15);
        }
        #pragma unroll
        for (int tj = 0; tj < 4; tj++) {
            const bf16* p = &ldsB[buf][(laneH * 8) * LDB_SH + nv * 64 + tj * 16 + laneN];
            #pragma unroll
            for (int s = 0; s < 16; s++) bfr[tj][s] = p[(s < 8 ? s : s + 8) * LDB_SH];
        }
        #pragma unroll
        for (int ti = 0; ti < 2; ti++)
            #pragma unroll
            for (int tj = 0; tj < 4; tj++)
                acc[ti][tj] = __builtin_amdgcn_wmma_f32_16x16x32_bf16(
                    false, af[ti], false, bfr[tj], (short)0, acc[ti][tj], false, false);
    };

    const int nK = K >> 5;
    stage(0, 0);
    tdmWait();
    __syncthreads();
    for (int kt = 0; kt < nK; kt++) {
        int cur = kt & 1;
        if (kt + 1 < nK) stage(kt + 1, cur ^ 1);   // async prefetch of next tile
        compute(cur);
        if (kt + 1 < nK) {
            tdmWait();
            __syncthreads();
        }
    }

    // ------------------------------- epilogue -------------------------------
    float nw0 = 0.f, nw1 = 0.f;
    if constexpr (EPI == 3) {
        float a0 = attnw[0], a1 = attnw[1];
        float mx = fmaxf(a0, a1);
        float t0 = (a0 - mx) * 0.1f + noisev[0] * 1e-6f;
        float t1 = (a1 - mx) * 0.1f + noisev[1] * 1e-6f;
        float e0 = __expf(t0), e1 = __expf(t1);
        float is = 1.f / (e0 + e1);
        nw0 = e0 * is; nw1 = e1 * is;
    }
    bf16*  outBf  = (EPI == 0 || EPI == 2) ? outBf0 + (size_t)z * zO : nullptr;
    float* outFin = (EPI == 3) ? outFin0 + (size_t)z * zF : nullptr;
    float  lm = -3.4e38f;

    #pragma unroll
    for (int ti = 0; ti < 2; ti++)
        #pragma unroll
        for (int tj = 0; tj < 4; tj++)
            #pragma unroll
            for (int v = 0; v < 8; v++) {
                int gm = mBase + mw * 32 + ti * 16 + laneH * 8 + v;
                int gn = nBase + nv * 64 + tj * 16 + laneN;
                float val = acc[ti][tj][v];
                if constexpr (EPI == 0) {
                    outBf[(size_t)gm * ldo + gn] = (bf16)(val + bias[gn]);
                } else if constexpr (EPI == 1) {
                    outF[(size_t)gm * ldo + gn] = val;
                    lm = fmaxf(lm, val);
                } else if constexpr (EPI == 2) {
                    float d2 = fmaxf(sq[gm] + sq[gn] - 2.f * val, 0.f);
                    outBf[(size_t)gm * ldo + gn] = (bf16)__expf(-d2 * scaleExp);
                } else {
                    outFin[(size_t)gm * ldo + gn] = nw0 * aggV[gn] + nw1 * (val * (1.f / 1024.f));
                }
            }

    if constexpr (EPI == 1) {
        #pragma unroll
        for (int o = 16; o > 0; o >>= 1) lm = fmaxf(lm, __shfl_xor(lm, o, 32));
        __shared__ float smax[8];
        if (lane == 0) smax[wv] = lm;
        __syncthreads();
        if (tid == 0) {
            float m = smax[0];
            for (int i = 1; i < 8; i++) m = fmaxf(m, smax[i]);
            atomicMax(gmaxKey, encodeKey(m));
        }
    }
}

// fp32 -> bf16 + per-row sum of squares (rows of 512)
__global__ __launch_bounds__(256)
void prep_features(const float* __restrict__ f, bf16* __restrict__ fb, float* __restrict__ sq)
{
    int row = blockIdx.x, tid = threadIdx.x;
    const float* src = f + (size_t)row * 512;
    bf16* dst = fb + (size_t)row * 512;
    float s = 0.f;
    #pragma unroll
    for (int j = tid; j < 512; j += 256) { float v = src[j]; dst[j] = (bf16)v; s += v * v; }
    #pragma unroll
    for (int o = 16; o > 0; o >>= 1) s += __shfl_xor(s, o, 32);
    __shared__ float red[8];
    if ((tid & 31) == 0) red[tid >> 5] = s;
    __syncthreads();
    if (tid == 0) {
        float t = 0.f;
        for (int i = 0; i < 8; i++) t += red[i];
        sq[row] = t;
    }
}

__global__ void f32_to_bf16(const float* __restrict__ s, bf16* __restrict__ d, int n)
{
    int i = blockIdx.x * 256 + threadIdx.x;
    if (i < n) d[i] = (bf16)s[i];
}

__global__ void init_kernel(float* __restrict__ cw, unsigned* __restrict__ gmaxKey)
{
    int t = blockIdx.x * 256 + threadIdx.x;
    if (t < 1024) cw[t] = 0.f;
    if (t == 0) *gmaxKey = 0x007FFFFFu;   // encodeKey(-inf)
}

// Row softmax of scores/gmax; accumulate column sums of softmax weights.
__global__ __launch_bounds__(256)
void softmax_cw(const float* __restrict__ scores, const unsigned* __restrict__ gmaxKey,
                float* __restrict__ cw)
{
    int r = blockIdx.x, tid = threadIdx.x;
    float inv = 1.0f / decodeKey(*gmaxKey);
    float v[4];
    #pragma unroll
    for (int j = 0; j < 4; j++) v[j] = scores[(size_t)r * 1024 + tid + j * 256] * inv;
    float m = fmaxf(fmaxf(v[0], v[1]), fmaxf(v[2], v[3]));
    #pragma unroll
    for (int o = 16; o > 0; o >>= 1) m = fmaxf(m, __shfl_xor(m, o, 32));
    __shared__ float red[8];
    __shared__ float bmax, bsum;
    int w = tid >> 5, l = tid & 31;
    if (l == 0) red[w] = m;
    __syncthreads();
    if (tid == 0) { float t = red[0]; for (int i = 1; i < 8; i++) t = fmaxf(t, red[i]); bmax = t; }
    __syncthreads();
    float M = bmax, s = 0.f;
    #pragma unroll
    for (int j = 0; j < 4; j++) { v[j] = __expf(v[j] - M); s += v[j]; }
    #pragma unroll
    for (int o = 16; o > 0; o >>= 1) s += __shfl_xor(s, o, 32);
    if (l == 0) red[w] = s;
    __syncthreads();
    if (tid == 0) { float t = 0.f; for (int i = 0; i < 8; i++) t += red[i]; bsum = t; }
    __syncthreads();
    float invS = 1.0f / bsum;
    #pragma unroll
    for (int j = 0; j < 4; j++) atomicAdd(&cw[tid + j * 256], v[j] * invS);
}

// agg[n] = (colsum_w @ v)[n] / 1024  == mean over rows of (w @ v)
__global__ __launch_bounds__(256)
void agg_kernel(const float* __restrict__ cw, const bf16* __restrict__ vmat,
                float* __restrict__ agg)
{
    int n = blockIdx.x * 256 + threadIdx.x;   // 0..511
    float s = 0.f;
    for (int k = 0; k < 1024; k++) s += cw[k] * (float)vmat[(size_t)k * 512 + n];
    agg[n] = s * (1.0f / 1024.0f);
}

extern "C" void kernel_launch(void* const* d_in, const int* in_sizes, int n_in,
                              void* d_out, int out_size, void* d_ws, size_t ws_size,
                              hipStream_t stream)
{
    const float* features = (const float*)d_in[0];
    const float* Wq = (const float*)d_in[1];
    const float* bq = (const float*)d_in[2];
    const float* Wk = (const float*)d_in[3];
    const float* bk = (const float*)d_in[4];
    const float* Wv = (const float*)d_in[5];
    const float* bv = (const float*)d_in[6];
    const float* attnw = (const float*)d_in[7];
    const float* noise = (const float*)d_in[8];
    const int*   sidx  = (const int*)d_in[9];
    float* out = (float*)d_out;

    const int N = 131072, D = 512, S = 1024, NB = 128;

    char* base = (char*)d_ws;
    size_t off = 0;
    auto alloc = [&](size_t bytes) -> void* {
        void* p = base + off;
        off = (off + bytes + 255) & ~(size_t)255;
        return p;
    };
    bf16*  fbf    = (bf16*)alloc((size_t)N * D * 2);
    float* dsq    = (float*)alloc((size_t)N * 4);
    bf16*  wqb    = (bf16*)alloc((size_t)D * D * 2);
    bf16*  wkb    = (bf16*)alloc((size_t)D * D * 2);
    bf16*  wvb    = (bf16*)alloc((size_t)D * D * 2);
    bf16*  qb     = (bf16*)alloc((size_t)S * D * 2);
    bf16*  kb     = (bf16*)alloc((size_t)S * D * 2);
    bf16*  vb     = (bf16*)alloc((size_t)S * D * 2);
    float* scores = (float*)alloc((size_t)S * S * 4);
    float* cw     = (float*)alloc((size_t)S * 4);
    float* aggv   = (float*)alloc((size_t)D * 4);
    unsigned* gmax = (unsigned*)alloc(256);
    size_t used = off;

    const size_t wblk = (size_t)S * S * 2;   // 2 MB bf16 W per Gaussian block
    int chunk = 1;
    if (ws_size > used + wblk) {
        size_t c = (ws_size - used) / wblk;
        chunk = (int)(c > (size_t)NB ? (size_t)NB : c);
        if (chunk < 1) chunk = 1;
    }
    bf16* Wbuf = (bf16*)(base + used);

    // ---- preprocessing ----
    prep_features<<<N, 256, 0, stream>>>(features, fbf, dsq);
    f32_to_bf16<<<(D * D + 255) / 256, 256, 0, stream>>>(Wq, wqb, D * D);
    f32_to_bf16<<<(D * D + 255) / 256, 256, 0, stream>>>(Wk, wkb, D * D);
    f32_to_bf16<<<(D * D + 255) / 256, 256, 0, stream>>>(Wv, wvb, D * D);
    init_kernel<<<4, 256, 0, stream>>>(cw, gmax);

    // ---- sampled attention: q/k/v = features[idx] @ W^T + b ----
    dim3 gqkv(S / 64, D / 256, 1);
    gemm_wmma<0, true, true><<<gqkv, 256, 0, stream>>>(
        fbf, wqb, D, D, D, 0, 0, sidx, bq, qb, D, 0,
        nullptr, nullptr, nullptr, 0, 0.f, nullptr, nullptr, nullptr, nullptr, 0);
    gemm_wmma<0, true, true><<<gqkv, 256, 0, stream>>>(
        fbf, wkb, D, D, D, 0, 0, sidx, bk, kb, D, 0,
        nullptr, nullptr, nullptr, 0, 0.f, nullptr, nullptr, nullptr, nullptr, 0);
    gemm_wmma<0, true, true><<<gqkv, 256, 0, stream>>>(
        fbf, wvb, D, D, D, 0, 0, sidx, bv, vb, D, 0,
        nullptr, nullptr, nullptr, 0, 0.f, nullptr, nullptr, nullptr, nullptr, 0);

    // scores = q @ k^T (f32) + global max
    dim3 gsc(S / 64, S / 256, 1);
    gemm_wmma<1, true, false><<<gsc, 256, 0, stream>>>(
        qb, kb, D, D, D, 0, 0, nullptr, nullptr, nullptr, S, 0,
        scores, gmax, nullptr, 0, 0.f, nullptr, nullptr, nullptr, nullptr, 0);

    softmax_cw<<<S, 256, 0, stream>>>(scores, gmax, cw);
    agg_kernel<<<2, 256, 0, stream>>>(cw, vb, aggv);

    // ---- Gaussian block attention + final combine ----
    for (int b0 = 0; b0 < NB; b0 += chunk) {
        int nz = (chunk < NB - b0) ? chunk : (NB - b0);
        const bf16* xblk = fbf + (size_t)b0 * S * D;
        dim3 g1(S / 64, S / 256, nz);
        gemm_wmma<2, true, false><<<g1, 256, 0, stream>>>(
            xblk, xblk, D, D, D, (size_t)S * D, (size_t)S * D, nullptr, nullptr,
            Wbuf, S, (size_t)S * S, nullptr, nullptr,
            dsq + (size_t)b0 * S, (size_t)S, 1.0f / 102.4f,
            nullptr, nullptr, nullptr, nullptr, 0);
        dim3 g2(S / 64, D / 256, nz);
        gemm_wmma<3, false, false><<<g2, 256, 0, stream>>>(
            Wbuf, xblk, S, D, S, (size_t)S * S, (size_t)S * D, nullptr, nullptr,
            nullptr, D, 0, nullptr, nullptr, nullptr, 0, 0.f,
            aggv, attnw, noise, out + (size_t)b0 * S * D, (size_t)S * D);
    }
}
